// LiTrDecoder_987842478257
// MI455X (gfx1250) — matrix operand order repaired and checked by reading, not compiled, not run on hardware
//
#include <hip/hip_runtime.h>
#include <hip/hip_bf16.h>

// ---------------- problem constants ----------------
#define D_    256
#define NH_   8
#define DH_   32
#define LV_   3
#define NPT_  4
#define NL_   6
#define NQ_   300
#define BB_   8
#define NC_   80
#define DFF_  1024
#define S_    8400
#define ROWS_ (BB_ * NQ_)   // 2400
#define NQP_  320           // NQ padded to multiple of 32 for WMMA K

typedef __attribute__((ext_vector_type(16))) _Float16 v16h;
typedef __attribute__((ext_vector_type(8)))  _Float16 h8;
typedef __attribute__((ext_vector_type(4)))  _Float16 h4;
typedef __attribute__((ext_vector_type(8)))  float    v8f;
typedef __attribute__((ext_vector_type(4)))  float    f4;

__device__ __forceinline__ float geluf(float x) {
  return 0.5f * x * (1.0f + erff(x * 0.70710678118654752f));
}
__device__ __forceinline__ float wave_sum(float v) {
#pragma unroll
  for (int o = 16; o > 0; o >>= 1) v += __shfl_xor(v, o, 32);
  return v;
}
__device__ __forceinline__ float wave_max(float v) {
#pragma unroll
  for (int o = 16; o > 0; o >>= 1) v = fmaxf(v, __shfl_xor(v, o, 32));
  return v;
}
__device__ __forceinline__ h4 cvt4(f4 v) {
  h4 r;
  r[0] = (_Float16)v[0]; r[1] = (_Float16)v[1];
  r[2] = (_Float16)v[2]; r[3] = (_Float16)v[3];
  return r;
}

// ---- WMMA fragment gathers (ISA 7.12.2, 16-bit, wave32) ----
// A tile in LDS: row-major [row][k], ld = 32 halfs. Per-lane fragment is two
// contiguous 16B chunks -> two ds_load_b128.
__device__ __forceinline__ v16h load_a_frag32(const _Float16* sm, int row0) {
  const int lane = threadIdx.x & 31;
  const int m = lane & 15;
  const int lo = (lane & 16) ? 16 : 0;
  const _Float16* p = sm + (row0 + m) * 32 + lo;
  h8 a0 = *(const h8*)(p);        // K = lo+0 .. lo+7
  h8 a1 = *(const h8*)(p + 8);    // K = lo+8 .. lo+15
  v16h a;
#pragma unroll
  for (int e = 0; e < 8; ++e) { a[e] = a0[e]; a[e + 8] = a1[e]; }
  return a;
}
// B tile in LDS: COLUMN-major [col][k], 32 k-halfs per column. Per-lane fragment
// (k = kOff..kOff+15, fixed n) is 32 contiguous, 32B-aligned bytes -> ds_load_b128 x2.
__device__ __forceinline__ v16h load_b_fragT(const _Float16* sm, int col0) {
  const int lane = threadIdx.x & 31;
  const int n = lane & 15;
  const int ko = (lane & 16) ? 16 : 0;
  return *(const v16h*)(sm + (col0 + n) * 32 + ko);
}

// ---------------- generic WMMA GEMM ----------------
// C[M,N](ldc) = act( (A + A2?)[M,K] @ B[K,N](ldb) + bias[N] ),  A contiguous lda==K.
// Requires K % 32 == 0 and all pointers/strides 4-float aligned (true for all uses).
// block = 128 threads (4 waves), 64x64 C tile, wave -> 32x32 (2x2 wmma).
__global__ __launch_bounds__(128) void gemm_wmma(
    const float* __restrict__ A, const float* __restrict__ A2,
    const float* __restrict__ Bm, int ldb,
    const float* __restrict__ bias,
    float* __restrict__ C, int ldc,
    int M, int N, int K, int act) {
  __shared__ _Float16 sA[64 * 32];   // [row][k]
  __shared__ _Float16 sB[64 * 32];   // [col][k]
  const int tM = blockIdx.y * 64, tN = blockIdx.x * 64;
  const int wave = threadIdx.x >> 5;
  const int wm = (wave >> 1) * 32, wn = (wave & 1) * 32;
  v8f acc[2][2] = {};
  for (int k0 = 0; k0 < K; k0 += 32) {
    // A tile: 64x32, float4 loads, h4 stores
    for (int i = threadIdx.x * 4; i < 64 * 32; i += 512) {
      int r = i >> 5, c = i & 31;
      int gr = tM + r;
      f4 av = {0.f, 0.f, 0.f, 0.f};
      if (gr < M) {
        av = *(const f4*)(A + (size_t)gr * K + k0 + c);
        if (A2) av += *(const f4*)(A2 + (size_t)gr * K + k0 + c);
      }
      *(h4*)(sA + r * 32 + c) = cvt4(av);
    }
    // B tile: 32x64 -> column-major LDS
    for (int i = threadIdx.x * 4; i < 32 * 64; i += 512) {
      int r = i >> 6, c = i & 63;   // c multiple of 4
      int gk = k0 + r, gc = tN + c;
      f4 bv = {0.f, 0.f, 0.f, 0.f};
      if (gc + 3 < N) {
        bv = *(const f4*)(Bm + (size_t)gk * ldb + gc);
      } else {
#pragma unroll
        for (int t = 0; t < 4; ++t)
          if (gc + t < N) bv[t] = Bm[(size_t)gk * ldb + gc + t];
      }
#pragma unroll
      for (int t = 0; t < 4; ++t) sB[(c + t) * 32 + r] = (_Float16)bv[t];
    }
    __syncthreads();
#pragma unroll
    for (int i = 0; i < 2; ++i) {
      v16h af = load_a_frag32(sA, wm + 16 * i);
#pragma unroll
      for (int j = 0; j < 2; ++j) {
        v16h bf = load_b_fragT(sB, wn + 16 * j);
        acc[i][j] = __builtin_amdgcn_wmma_f32_16x16x32_f16(
            false, af, false, bf, (short)0, acc[i][j], false, false);
      }
    }
    __syncthreads();
  }
  const int lane = threadIdx.x & 31;
  const int n = lane & 15;
  const int mo = (lane & 16) ? 8 : 0;
#pragma unroll
  for (int i = 0; i < 2; ++i)
#pragma unroll
    for (int j = 0; j < 2; ++j)
#pragma unroll
      for (int r = 0; r < 8; ++r) {
        int gm = tM + wm + 16 * i + mo + r;
        int gn = tN + wn + 16 * j + n;
        if (gm < M && gn < N) {
          float vv = acc[i][j][r];
          if (bias) vv += bias[gn];
          if (act == 1) vv = geluf(vv);
          C[(size_t)gm * ldc + gn] = vv;
        }
      }
}

// ---------------- attention scores: per (b,h) Q(300,32) @ K^T ----------------
__global__ __launch_bounds__(128) void attn_scores(const float* __restrict__ qk,
                                                   float* __restrict__ sc) {
  __shared__ _Float16 sA[64 * 32];   // [qrow][d]
  __shared__ _Float16 sB[64 * 32];   // [krow][d]  (== column-major of K^T)
  const int z = blockIdx.z, b = z >> 3, h = z & 7;
  const float* q = qk + (size_t)b * NQ_ * 512 + h * DH_;
  const float* k = qk + (size_t)b * NQ_ * 512 + 256 + h * DH_;
  float* out = sc + (size_t)z * NQ_ * NQP_;
  const int ti = blockIdx.y * 64, tj = blockIdx.x * 64;
  for (int i = threadIdx.x * 4; i < 64 * 32; i += 512) {
    int r = i >> 5, c = i & 31;
    f4 v = {0.f, 0.f, 0.f, 0.f};
    if (ti + r < NQ_) v = *(const f4*)(q + (size_t)(ti + r) * 512 + c);
    *(h4*)(sA + r * 32 + c) = cvt4(v);
  }
  for (int i = threadIdx.x * 4; i < 64 * 32; i += 512) {
    int j = i >> 5, d = i & 31;
    f4 v = {0.f, 0.f, 0.f, 0.f};
    if (tj + j < NQ_) v = *(const f4*)(k + (size_t)(tj + j) * 512 + d);
    *(h4*)(sB + j * 32 + d) = cvt4(v);
  }
  __syncthreads();
  const int wave = threadIdx.x >> 5;
  const int wm = (wave >> 1) * 32, wn = (wave & 1) * 32;
  v8f acc[2][2] = {};
#pragma unroll
  for (int i = 0; i < 2; ++i) {
    v16h af = load_a_frag32(sA, wm + 16 * i);
#pragma unroll
    for (int j = 0; j < 2; ++j) {
      v16h bf = load_b_fragT(sB, wn + 16 * j);
      acc[i][j] = __builtin_amdgcn_wmma_f32_16x16x32_f16(
          false, af, false, bf, (short)0, acc[i][j], false, false);
    }
  }
  const int lane = threadIdx.x & 31;
  const int n = lane & 15;
  const int mo = (lane & 16) ? 8 : 0;
#pragma unroll
  for (int i = 0; i < 2; ++i)
#pragma unroll
    for (int j = 0; j < 2; ++j)
#pragma unroll
      for (int r = 0; r < 8; ++r) {
        int gm = ti + wm + 16 * i + mo + r;
        int gn = tj + wn + 16 * j + n;
        if (gm < NQ_ && gn < NQ_) out[(size_t)gm * NQP_ + gn] = acc[i][j][r];
      }
}

// softmax over rows of 300 (with 1/sqrt(32) scale), zero pad cols 300..319
__global__ void attn_softmax(float* __restrict__ sc) {
  const int wid = blockIdx.x * 4 + (threadIdx.x >> 5);
  if (wid >= BB_ * NH_ * NQ_) return;
  const int lane = threadIdx.x & 31;
  float* row = sc + (size_t)(wid / NQ_) * NQ_ * NQP_ + (size_t)(wid % NQ_) * NQP_;
  const float scale = 0.17677669529663687f;
  float m = -1e30f;
  for (int j = lane; j < NQ_; j += 32) m = fmaxf(m, row[j] * scale);
  m = wave_max(m);
  float s = 0.f;
  for (int j = lane; j < NQ_; j += 32) s += expf(row[j] * scale - m);
  s = wave_sum(s);
  const float inv = 1.f / s;
  for (int j = lane; j < NQP_; j += 32)
    row[j] = (j < NQ_) ? expf(row[j] * scale - m) * inv : 0.f;
}

// O(300,32) = att(300,320) @ V(300/320,32) per (b,h); K loop over 320
__global__ __launch_bounds__(128) void attn_av(const float* __restrict__ sc,
                                               const float* __restrict__ v,
                                               float* __restrict__ o) {
  __shared__ _Float16 sA[64 * 32];   // [row][k]
  __shared__ _Float16 sB[32 * 32];   // [col][k]
  const int z = blockIdx.y, b = z >> 3, h = z & 7;
  const float* att = sc + (size_t)z * NQ_ * NQP_;
  const float* vb = v + (size_t)b * NQ_ * D_ + h * DH_;
  float* ob = o + (size_t)b * NQ_ * D_ + h * DH_;
  const int ti = blockIdx.x * 64;
  const int wave = threadIdx.x >> 5;
  v8f acc[2] = {};
  for (int k0 = 0; k0 < NQP_; k0 += 32) {
    for (int i = threadIdx.x * 4; i < 64 * 32; i += 512) {
      int r = i >> 5, c = i & 31;
      f4 av = {0.f, 0.f, 0.f, 0.f};
      if (ti + r < NQ_) av = *(const f4*)(att + (size_t)(ti + r) * NQP_ + k0 + c);
      *(h4*)(sA + r * 32 + c) = cvt4(av);
    }
    for (int i = threadIdx.x * 4; i < 32 * 32; i += 512) {
      int r = i >> 5, c = i & 31;
      f4 vv = {0.f, 0.f, 0.f, 0.f};
      if (k0 + r < NQ_) vv = *(const f4*)(vb + (size_t)(k0 + r) * D_ + c);
#pragma unroll
      for (int t = 0; t < 4; ++t) sB[(c + t) * 32 + r] = (_Float16)vv[t];
    }
    __syncthreads();
    v16h af = load_a_frag32(sA, wave * 16);
#pragma unroll
    for (int j = 0; j < 2; ++j) {
      v16h bf = load_b_fragT(sB, 16 * j);
      acc[j] = __builtin_amdgcn_wmma_f32_16x16x32_f16(
          false, af, false, bf, (short)0, acc[j], false, false);
    }
    __syncthreads();
  }
  const int lane = threadIdx.x & 31;
  const int n = lane & 15;
  const int mo = (lane & 16) ? 8 : 0;
#pragma unroll
  for (int j = 0; j < 2; ++j)
#pragma unroll
    for (int r = 0; r < 8; ++r) {
      int gm = ti + wave * 16 + mo + r;
      if (gm < NQ_) ob[(size_t)gm * D_ + 16 * j + n] = acc[j][r];
    }
}

// x = LN(x + res) * g + b, one wave per 256-wide row
__global__ void add_ln(float* __restrict__ x, const float* __restrict__ res,
                       const float* __restrict__ g, const float* __restrict__ bta) {
  const int row = blockIdx.x * 4 + (threadIdx.x >> 5);
  if (row >= ROWS_) return;
  const int lane = threadIdx.x & 31;
  float vv[8];
  float s = 0.f;
#pragma unroll
  for (int e = 0; e < 8; ++e) {
    int c = lane + 32 * e;
    vv[e] = x[(size_t)row * D_ + c] + res[(size_t)row * D_ + c];
    s += vv[e];
  }
  const float mean = wave_sum(s) * (1.f / D_);
  float q = 0.f;
#pragma unroll
  for (int e = 0; e < 8; ++e) { float d = vv[e] - mean; q += d * d; }
  const float inv = rsqrtf(wave_sum(q) * (1.f / D_) + 1e-5f);
#pragma unroll
  for (int e = 0; e < 8; ++e) {
    int c = lane + 32 * e;
    x[(size_t)row * D_ + c] = (vv[e] - mean) * inv * g[c] + bta[c];
  }
}

__global__ void sigmoid_k(const float* __restrict__ x, float* __restrict__ y, int n) {
  int i = blockIdx.x * blockDim.x + threadIdx.x;
  if (i < n) y[i] = 1.f / (1.f + expf(-x[i]));
}

// qgelu = gelu(ref_det(2400,4) @ qp_w1(4,512) + qp_b1)
__global__ void qpos_first(const float* __restrict__ refd, const float* __restrict__ w1,
                           const float* __restrict__ b1, float* __restrict__ out512) {
  int idx = blockIdx.x * 256 + threadIdx.x;
  if (idx >= ROWS_ * 512) return;
  int row = idx >> 9, col = idx & 511;
  float s = b1[col];
#pragma unroll
  for (int c = 0; c < 4; ++c) s += refd[row * 4 + c] * w1[c * 512 + col];
  out512[idx] = geluf(s);
}

// softmax over groups of LV*NPT=12 within rows of 96
__global__ void deform_softmax(float* __restrict__ aw) {
  int idx = blockIdx.x * blockDim.x + threadIdx.x;
  if (idx >= ROWS_ * NH_) return;
  float* p = aw + (size_t)(idx / NH_) * 96 + (idx % NH_) * 12;
  float m = -1e30f;
  for (int j = 0; j < 12; ++j) m = fmaxf(m, p[j]);
  float e[12], s = 0.f;
  for (int j = 0; j < 12; ++j) { e[j] = expf(p[j] - m); s += e[j]; }
  const float inv = 1.f / s;
  for (int j = 0; j < 12; ++j) p[j] = e[j] * inv;
}

// deformable bilinear gather: one wave per (b,q,h), lane = channel d
__global__ void deform_sample(const float* __restrict__ val, const float* __restrict__ refd,
                              const float* __restrict__ offb, const float* __restrict__ awb,
                              float* __restrict__ res) {
  const int wid = blockIdx.x * 4 + (threadIdx.x >> 5);
  if (wid >= ROWS_ * NH_) return;
  const int lane = threadIdx.x & 31;
  const int h = wid % NH_, bq = wid / NH_, b = bq / NQ_;
  const float* rp = refd + (size_t)bq * 4;
  const float rx = rp[0], ry = rp[1], rw = rp[2], rh = rp[3];
  const int hls[3] = {80, 40, 20}, wls[3] = {80, 40, 20}, sts[3] = {0, 6400, 8000};
  float acc = 0.f;
  for (int lvl = 0; lvl < LV_; ++lvl) {
    const int hl = hls[lvl], wl = wls[lvl], st = sts[lvl];
    for (int pt = 0; pt < NPT_; ++pt) {
      const int fi = (h * LV_ + lvl) * NPT_ + pt;
      const float ox = offb[(size_t)bq * 192 + fi * 2];
      const float oy = offb[(size_t)bq * 192 + fi * 2 + 1];
      const float a = awb[(size_t)bq * 96 + fi];
      const float x = (rx + ox * 0.125f * rw) * (float)wl - 0.5f;
      const float y = (ry + oy * 0.125f * rh) * (float)hl - 0.5f;
      const float x0 = floorf(x), y0 = floorf(y);
      const float lx = x - x0, ly = y - y0;
      const float wts[4] = {(1.f - lx) * (1.f - ly), lx * (1.f - ly),
                            (1.f - lx) * ly, lx * ly};
      const float xs[4] = {x0, x0 + 1.f, x0, x0 + 1.f};
      const float ys[4] = {y0, y0, y0 + 1.f, y0 + 1.f};
#pragma unroll
      for (int c2 = 0; c2 < 4; ++c2) {
        const float xi = xs[c2], yi = ys[c2];
        if (xi >= 0.f && xi <= (float)(wl - 1) && yi >= 0.f && yi <= (float)(hl - 1)) {
          const int s = st + (int)yi * wl + (int)xi;
          acc += a * wts[c2] * val[((size_t)b * S_ + s) * D_ + h * DH_ + lane];
        }
      }
    }
  }
  res[(size_t)bq * D_ + h * DH_ + lane] = acc;
}

// bboxes[i] == inter (ref_points == ref_det every layer); update refd in place
__global__ void bbox_update(const float* __restrict__ delta, float* __restrict__ refd,
                            float* __restrict__ outBox) {
  int idx = blockIdx.x * blockDim.x + threadIdx.x;
  if (idx >= ROWS_) return;
#pragma unroll
  for (int c = 0; c < 4; ++c) {
    float r = fminf(fmaxf(refd[idx * 4 + c], 0.f), 1.f);
    float inv = logf(fmaxf(r, 1e-5f) / fmaxf(1.f - r, 1e-5f));
    float it = 1.f / (1.f + expf(-(delta[idx * 4 + c] + inv)));
    outBox[(size_t)idx * 84 + c] = it;
    refd[idx * 4 + c] = it;
  }
}

// ---------------- host orchestration ----------------
extern "C" void kernel_launch(void* const* d_in, const int* in_sizes, int n_in,
                              void* d_out, int out_size, void* d_ws, size_t ws_size,
                              hipStream_t stream) {
  const float* target  = (const float*)d_in[0];
  const float* rpu     = (const float*)d_in[1];
  const float* memory  = (const float*)d_in[2];
  const float* qp_w1   = (const float*)d_in[3];
  const float* qp_b1   = (const float*)d_in[4];
  const float* qp_w2   = (const float*)d_in[5];
  const float* qp_b2   = (const float*)d_in[6];
  const float* sa_w    = (const float*)d_in[7];
  const float* sa_b    = (const float*)d_in[8];
  const float* sa_wo   = (const float*)d_in[9];
  const float* sa_bo   = (const float*)d_in[10];
  const float* ca_w_off  = (const float*)d_in[11];
  const float* ca_b_off  = (const float*)d_in[12];
  const float* ca_w_attn = (const float*)d_in[13];
  const float* ca_b_attn = (const float*)d_in[14];
  const float* ca_w_val  = (const float*)d_in[15];
  const float* ca_b_val  = (const float*)d_in[16];
  const float* ca_w_out  = (const float*)d_in[17];
  const float* ca_b_out  = (const float*)d_in[18];
  const float* ln1_g = (const float*)d_in[19];
  const float* ln1_b = (const float*)d_in[20];
  const float* ln2_g = (const float*)d_in[21];
  const float* ln2_b = (const float*)d_in[22];
  const float* ln3_g = (const float*)d_in[23];
  const float* ln3_b = (const float*)d_in[24];
  const float* ff_w1 = (const float*)d_in[25];
  const float* ff_b1 = (const float*)d_in[26];
  const float* ff_w2 = (const float*)d_in[27];
  const float* ff_b2 = (const float*)d_in[28];
  const float* bb_w1 = (const float*)d_in[29];
  const float* bb_b1 = (const float*)d_in[30];
  const float* bb_w2 = (const float*)d_in[31];
  const float* bb_b2 = (const float*)d_in[32];
  const float* bb_w3 = (const float*)d_in[33];
  const float* bb_b3 = (const float*)d_in[34];
  const float* sc_w  = (const float*)d_in[35];
  const float* sc_b  = (const float*)d_in[36];
  float* out_all = (float*)d_out;

  // workspace layout (floats), ~133 MB total
  float* ws   = (float*)d_ws;
  float* out  = ws;                    // 2400*256
  float* qpos = out  + ROWS_ * D_;
  float* qgel = qpos + ROWS_ * D_;     // 2400*512
  float* qk   = qgel + ROWS_ * 512;    // 2400*512
  float* vbuf = qk   + ROWS_ * 512;    // 2400*256
  float* scb  = vbuf + ROWS_ * D_;     // 64*300*320
  float* obuf = scb  + (size_t)BB_ * NH_ * NQ_ * NQP_;
  float* t2   = obuf + ROWS_ * D_;
  float* valb = t2   + ROWS_ * D_;     // 67200*256
  float* offb = valb + (size_t)BB_ * S_ * D_;   // 2400*192
  float* awb  = offb + ROWS_ * 192;    // 2400*96
  float* ffh  = awb  + ROWS_ * 96;     // 2400*1024
  float* h1   = ffh  + ROWS_ * DFF_;
  float* h2   = h1   + ROWS_ * D_;
  float* delt = h2   + ROWS_ * D_;     // 2400*4
  float* refd = delt + ROWS_ * 4;      // 2400*4

  auto gemm = [&](const float* A, const float* A2, const float* Bm, int ldb,
                  const float* bias, float* C, int ldc, int M, int N, int K, int act) {
    dim3 g((N + 63) / 64, (M + 63) / 64);
    gemm_wmma<<<g, 128, 0, stream>>>(A, A2, Bm, ldb, bias, C, ldc, M, N, K, act);
  };

  // init: out = target ; refd = sigmoid(ref_points_unact)
  hipMemcpyAsync(out, target, (size_t)ROWS_ * D_ * sizeof(float),
                 hipMemcpyDeviceToDevice, stream);
  sigmoid_k<<<(ROWS_ * 4 + 255) / 256, 256, 0, stream>>>(rpu, refd, ROWS_ * 4);

  for (int i = 0; i < NL_; ++i) {
    const float* sa_w_i  = sa_w  + (size_t)i * D_ * 3 * D_;
    const float* sa_b_i  = sa_b  + (size_t)i * 3 * D_;
    const float* sa_wo_i = sa_wo + (size_t)i * D_ * D_;
    const float* sa_bo_i = sa_bo + (size_t)i * D_;
    const float* w_off_i = ca_w_off  + (size_t)i * D_ * 192;
    const float* b_off_i = ca_b_off  + (size_t)i * 192;
    const float* w_att_i = ca_w_attn + (size_t)i * D_ * 96;
    const float* b_att_i = ca_b_attn + (size_t)i * 96;
    const float* w_val_i = ca_w_val  + (size_t)i * D_ * D_;
    const float* b_val_i = ca_b_val  + (size_t)i * D_;
    const float* w_out_i = ca_w_out  + (size_t)i * D_ * D_;
    const float* b_out_i = ca_b_out  + (size_t)i * D_;

    // qpos = gelu(refd @ qp_w1 + b1) @ qp_w2 + b2
    qpos_first<<<(ROWS_ * 512 + 255) / 256, 256, 0, stream>>>(refd, qp_w1, qp_b1, qgel);
    gemm(qgel, nullptr, qp_w2, D_, qp_b2, qpos, D_, ROWS_, D_, 512, 0);

    // self-attention: qk from (out+qpos), v from out
    gemm(out, qpos, sa_w_i, 3 * D_, sa_b_i, qk, 512, ROWS_, 512, D_, 0);
    gemm(out, nullptr, sa_w_i + 2 * D_, 3 * D_, sa_b_i + 2 * D_, vbuf, D_, ROWS_, D_, D_, 0);
    attn_scores<<<dim3(5, 5, BB_ * NH_), 128, 0, stream>>>(qk, scb);
    attn_softmax<<<(BB_ * NH_ * NQ_ + 3) / 4, 128, 0, stream>>>(scb);
    attn_av<<<dim3(5, BB_ * NH_), 128, 0, stream>>>(scb, vbuf, obuf);
    gemm(obuf, nullptr, sa_wo_i, D_, sa_bo_i, t2, D_, ROWS_, D_, D_, 0);
    add_ln<<<ROWS_ / 4, 128, 0, stream>>>(out, t2, ln1_g + i * D_, ln1_b + i * D_);

    // deformable cross-attention
    gemm(memory, nullptr, w_val_i, D_, b_val_i, valb, D_, BB_ * S_, D_, D_, 0);
    gemm(out, qpos, w_off_i, 192, b_off_i, offb, 192, ROWS_, 192, D_, 0);
    gemm(out, qpos, w_att_i, 96, b_att_i, awb, 96, ROWS_, 96, D_, 0);
    deform_softmax<<<(ROWS_ * NH_ + 127) / 128, 128, 0, stream>>>(awb);
    deform_sample<<<(ROWS_ * NH_ + 3) / 4, 128, 0, stream>>>(valb, refd, offb, awb, obuf);
    gemm(obuf, nullptr, w_out_i, D_, b_out_i, t2, D_, ROWS_, D_, D_, 0);
    add_ln<<<ROWS_ / 4, 128, 0, stream>>>(out, t2, ln2_g + i * D_, ln2_b + i * D_);

    // FFN
    gemm(out, nullptr, ff_w1 + (size_t)i * D_ * DFF_, DFF_, ff_b1 + (size_t)i * DFF_,
         ffh, DFF_, ROWS_, DFF_, D_, 1);
    gemm(ffh, nullptr, ff_w2 + (size_t)i * DFF_ * D_, D_, ff_b2 + (size_t)i * D_,
         t2, D_, ROWS_, D_, DFF_, 0);
    add_ln<<<ROWS_ / 4, 128, 0, stream>>>(out, t2, ln3_g + i * D_, ln3_b + i * D_);

    // heads
    gemm(out, nullptr, bb_w1 + (size_t)i * D_ * D_, D_, bb_b1 + (size_t)i * D_,
         h1, D_, ROWS_, D_, D_, 1);
    gemm(h1, nullptr, bb_w2 + (size_t)i * D_ * D_, D_, bb_b2 + (size_t)i * D_,
         h2, D_, ROWS_, D_, D_, 1);
    gemm(h2, nullptr, bb_w3 + (size_t)i * D_ * 4, 4, bb_b3 + (size_t)i * 4,
         delt, 4, ROWS_, 4, D_, 0);
    gemm(out, nullptr, sc_w + (size_t)i * D_ * NC_, NC_, sc_b + (size_t)i * NC_,
         out_all + (size_t)i * ROWS_ * 84 + 4, 84, ROWS_, NC_, D_, 0);
    bbox_update<<<(ROWS_ + 127) / 128, 128, 0, stream>>>(
        delt, refd, out_all + (size_t)i * ROWS_ * 84);
  }
}
